// SPD_82523501626162
// MI455X (gfx1250) — compile-verified, matrix operand order checked by reading, not run on hardware
//
#include <hip/hip_runtime.h>
#include <hip/hip_bf16.h>

// ---------------------------------------------------------------------------
// Types for CDNA5 WMMA
// ---------------------------------------------------------------------------
typedef __attribute__((ext_vector_type(16))) __bf16 v16bf;
typedef __attribute__((ext_vector_type(8)))  float  v8f;

struct U32x8 { uint4 a, b; };

__device__ __forceinline__ unsigned int f2bf(float f) {
    unsigned int u = __builtin_bit_cast(unsigned int, f);
    unsigned int r = u + 0x7FFFu + ((u >> 16) & 1u);   // round-to-nearest-even
    return r >> 16;
}
__device__ __forceinline__ float bf2f(unsigned short s) {
    return __builtin_bit_cast(float, (unsigned int)s << 16);
}
__device__ __forceinline__ v16bf frag16(uint4 x, uint4 y) {
    U32x8 u; u.a = x; u.b = y;
    return __builtin_bit_cast(v16bf, u);
}

// ---------------------------------------------------------------------------
// WMMA GEMM, LDS-free:
//   activations Xt: bf16, TRANSPOSED [b][col][c], channel stride Cp (mult of 32)
//   weights WP:     pre-packed uint fragments [ks][sr][lane][dword]
//   Y = act( W*X + bias (bn affine) (+ resT) ), written as transposed bf16 Yt
//   and/or plain f32 Yf (reference layout), with optional extra ReLU on Yt.
//
// Fragment mapping (ISA 7.12.2, 16-bit 16x32 operand), k-pair kp:
//   lane  = m | (((kp>>2)&1)<<4) ;  dword = (kp&3) | ((kp&8)?4:0)
// => for a lane (khalf = lane>>4) the 16 bf16 elements are exactly the two
//    contiguous 8-channel runs  [ks*32+khalf*8, +8)  and  [ks*32+16+khalf*8, +8)
//    of one column -> two global_load_b128 straight from the transposed tensor.
// D layout: lane holds 8 CONSECUTIVE output channels -> b128 store per subtile.
//
// Tile: 64 (O) x 128 (cols); 8 waves; wave (wr=wave&3, wcg=wave>>2) does one
// 16-row subtile across four 16-col subtiles -> 4 v_wmma per k-step.
// Waves whose row-subtile is beyond nSR compute on a clamped (valid) weight
// fragment and simply never store (oC >= Op early-out) -> branchless loop.
// ---------------------------------------------------------------------------
__global__ __launch_bounds__(256) void gemm_wmma2(
    const unsigned short* __restrict__ Xt,
    const unsigned int*  __restrict__ WP,
    const float* __restrict__ bias, const float* __restrict__ bnsc,
    const float* __restrict__ bnsh,
    const unsigned short* __restrict__ resT,
    unsigned short* __restrict__ Yt, float* __restrict__ Yf,
    int Cp, int O, int Op, int nSR,
    int colsIn, int colInBase, int colsOut, int colOutBase,
    int colCount, int NfOut, int actOut, int reluT)
{
    const int tid  = threadIdx.x;
    const int wave = tid >> 5, lane = tid & 31;
    const int b    = blockIdx.z;
    const int oBase = blockIdx.y * 64;
    const int cBase = blockIdx.x * 128;
    const int wr = wave & 3, wcg = wave >> 2;
    const int srG  = (oBase >> 4) + wr;
    const int srGl = srG < nSR ? srG : nSR - 1;   // clamp: loads always valid
    const int khalf = lane >> 4;
    const int nl    = lane & 15;
    const int nKS   = Cp >> 5;

    const unsigned short* Xb = Xt + (size_t)b * (unsigned)colsIn * (unsigned)Cp;

    int col[4];
#pragma unroll
    for (int t = 0; t < 4; ++t) {
        int c = cBase + (wcg * 4 + t) * 16 + nl;
        col[t] = c < colCount ? c : colCount - 1;
    }

    v8f acc[4];
#pragma unroll
    for (int t = 0; t < 4; ++t) acc[t] = (v8f){0.f,0.f,0.f,0.f,0.f,0.f,0.f,0.f};

    for (int ks = 0; ks < nKS; ++ks) {
        const uint4* pa = (const uint4*)(WP + ((((size_t)ks * nSR + srGl) << 5) + lane) * 8);
        uint4 a0 = pa[0];
        uint4 a1 = pa[1];
        v16bf af = frag16(a0, a1);
        const int kOff = (ks << 5) + khalf * 8;
        if (ks + 1 < nKS)   // global_prefetch_b8 for the next K slab
            __builtin_prefetch(Xb + (size_t)(unsigned)(colInBase + col[0]) * (unsigned)Cp + kOff + 32, 0, 1);
#pragma unroll
        for (int t = 0; t < 4; ++t) {
            const unsigned short* pb = Xb + (size_t)(unsigned)(colInBase + col[t]) * (unsigned)Cp + kOff;
            uint4 b0 = *(const uint4*)(pb);
            uint4 b1 = *(const uint4*)(pb + 16);
            v16bf bf = frag16(b0, b1);
            acc[t] = __builtin_amdgcn_wmma_f32_16x16x32_bf16(false, af, false, bf,
                                                             (short)0, acc[t], false, false);
        }
    }

    const int oC = oBase + wr * 16 + khalf * 8;     // first of 8 consecutive o
    if (oC >= Op) return;                           // also covers srG >= nSR
    const size_t outB = (size_t)b * (unsigned)colsOut * (unsigned)Op;
    const size_t yfB  = (size_t)b * (unsigned)O * (unsigned)NfOut;

#pragma unroll
    for (int t = 0; t < 4; ++t) {
        int c0 = cBase + (wcg * 4 + t) * 16 + nl;
        if (c0 >= colCount) continue;
        size_t colIdx = outB + (size_t)(unsigned)(colOutBase + c0) * (unsigned)Op + (unsigned)oC;
        float vv[8];
        if (resT) {
            uint4 rq = *(const uint4*)(resT + colIdx);
            unsigned int ru[4] = {rq.x, rq.y, rq.z, rq.w};
#pragma unroll
            for (int r = 0; r < 8; ++r)
                vv[r] = bf2f((unsigned short)((ru[r >> 1] >> ((r & 1) * 16)) & 0xffff));
        } else {
#pragma unroll
            for (int r = 0; r < 8; ++r) vv[r] = 0.f;
        }
#pragma unroll
        for (int r = 0; r < 8; ++r) {
            int o = oC + r;
            float v = acc[t][r];
            if (bias) v += bias[o];
            if (bnsc) v = v * bnsc[o] + bnsh[o];
            v += vv[r];
            if (actOut == 1)      v = fmaxf(v, 0.f);
            else if (actOut == 2) v = tanhf(v);
            vv[r] = (o < O) ? v : 0.f;
        }
        if (Yt) {
            uint4 u;
            unsigned int pk[4];
#pragma unroll
            for (int q = 0; q < 4; ++q) {
                float lo = reluT ? fmaxf(vv[2 * q],     0.f) : vv[2 * q];
                float hi = reluT ? fmaxf(vv[2 * q + 1], 0.f) : vv[2 * q + 1];
                pk[q] = (f2bf(hi) << 16) | f2bf(lo);
            }
            u.x = pk[0]; u.y = pk[1]; u.z = pk[2]; u.w = pk[3];
            *(uint4*)(Yt + colIdx) = u;       // global_store_b128
        }
        if (Yf) {
#pragma unroll
            for (int r = 0; r < 8; ++r)
                if (oC + r < O)
                    Yf[yfB + (size_t)(unsigned)(oC + r) * (unsigned)NfOut
                           + (unsigned)(colOutBase + c0)] = vv[r];
        }
    }
}

// ---------------------------------------------------------------------------
// Weight pre-pack: W (OxC f32) -> fragment-order uints [ks][sr][lane][dword]
// ---------------------------------------------------------------------------
__global__ void wpack_kernel(const float* __restrict__ W, unsigned int* __restrict__ out,
                             int C, int O, int nKS, int nSR)
{
    int total = nKS * nSR * 256;
    int j = blockIdx.x * 256 + threadIdx.x;
    if (j >= total) return;
    int d = j & 7, lane = (j >> 3) & 31, sk = j >> 8;
    int sr = sk % nSR, ks = sk / nSR;
    int m = lane & 15, khalf = lane >> 4;
    int kp = (d & 3) | (khalf << 2) | ((d & 4) ? 8 : 0);
    int k  = ks * 32 + kp * 2;
    int row = sr * 16 + m;
    int rs  = row < O ? row : O - 1;
    int k0  = k     < C ? k     : C - 1;
    int k1  = k + 1 < C ? k + 1 : C - 1;
    float x0 = W[(size_t)rs * C + k0];
    float x1 = W[(size_t)rs * C + k1];
    x0 = (row < O && k     < C) ? x0 : 0.f;
    x1 = (row < O && k + 1 < C) ? x1 : 0.f;
    out[j] = (f2bf(x1) << 16) | f2bf(x0);
}

// ---------------------------------------------------------------------------
// f32 channel-major (B,C,N) -> transposed padded bf16 (B,N,Cp)
// ---------------------------------------------------------------------------
__global__ void trpack_kernel(const float* __restrict__ src, unsigned short* __restrict__ dst,
                              int C, int Cp, int N)
{
    size_t i = (size_t)blockIdx.x * 256 + threadIdx.x;
    size_t total = (size_t)2 * N * Cp;
    if (i >= total) return;
    int c = (int)(i % Cp); size_t t = i / Cp; int col = (int)(t % N); int b = (int)(t / N);
    float v = (c < C) ? src[((size_t)b * C + c) * N + col] : 0.f;
    dst[i] = (unsigned short)f2bf(v);
}

// ---------------------------------------------------------------------------
// Brute-force KNN top-16 (including self), LDS-tiled candidates.
// ---------------------------------------------------------------------------
__global__ __launch_bounds__(256) void knn_kernel(const float* __restrict__ pos,
                                                  int* __restrict__ idx, int N)
{
    __shared__ float sx[256], sy[256], sz[256];
    const int b = blockIdx.y;
    const int q = blockIdx.x * 256 + threadIdx.x;
    const float* P = pos + (size_t)b * 3 * N;
    float qx = P[q], qy = P[N + q], qz = P[2 * N + q];
    float bd[16]; int bi[16];
#pragma unroll
    for (int j = 0; j < 16; ++j) { bd[j] = 3.0e38f; bi[j] = 0; }
    for (int t0 = 0; t0 < N; t0 += 256) {
        int i = t0 + threadIdx.x;
        sx[threadIdx.x] = P[i];
        sy[threadIdx.x] = P[N + i];
        sz[threadIdx.x] = P[2 * N + i];
        __syncthreads();
        for (int j = 0; j < 256; ++j) {
            float dx = qx - sx[j], dy = qy - sy[j], dz = qz - sz[j];
            float d = dx * dx + dy * dy + dz * dz;
            if (d < bd[15]) {
                bd[15] = d; bi[15] = t0 + j;
#pragma unroll
                for (int r = 15; r > 0; --r) {
                    if (bd[r] < bd[r - 1]) {
                        float td = bd[r]; bd[r] = bd[r - 1]; bd[r - 1] = td;
                        int   ti = bi[r]; bi[r] = bi[r - 1]; bi[r - 1] = ti;
                    }
                }
            }
        }
        __syncthreads();
    }
#pragma unroll
    for (int j = 0; j < 16; ++j)
        idx[((size_t)b * N + q) * 16 + j] = bi[j];
}

// ---------------------------------------------------------------------------
// Small helper kernels (transposed bf16 layouts)
// ---------------------------------------------------------------------------
__global__ void colmax_kernel(const unsigned short* __restrict__ xT, float* __restrict__ out,
                              int N, int Cp)
{
    int b = blockIdx.x, c = threadIdx.x;            // blockDim.x == Cp
    const unsigned short* p = xT + (size_t)b * N * Cp + c;
    float m = -3.0e38f;
    for (int i = 0; i < N; ++i) m = fmaxf(m, bf2f(p[(size_t)i * Cp]));
    out[b * Cp + c] = m;
}

__global__ void f1cat_kernel(const unsigned short* __restrict__ f128T, const float* __restrict__ gmax,
                             const float* __restrict__ fg, unsigned short* __restrict__ out, int N)
{
    size_t i = (size_t)blockIdx.x * 256 + threadIdx.x;     // out (B,N,768)
    size_t total = (size_t)2 * N * 768;
    if (i >= total) return;
    int c = (int)(i % 768); size_t t = i / 768; int col = (int)(t % N); int b = (int)(t / N);
    unsigned short v;
    if (c < 128)      v = f128T[((size_t)b * N + col) * 128 + c];
    else if (c < 256) v = (unsigned short)f2bf(gmax[b * 128 + (c - 128)]);
    else              v = (unsigned short)f2bf(fg[b * 512 + (c - 256)]);
    out[i] = v;
}

__global__ void kqcat_kernel(const unsigned short* __restrict__ keyT, const unsigned short* __restrict__ qT,
                             unsigned short* __restrict__ out, int N)
{
    size_t i = (size_t)blockIdx.x * 256 + threadIdx.x;     // out (B,N,256)
    size_t total = (size_t)2 * N * 256;
    if (i >= total) return;
    int c = (int)(i % 256); size_t t = i / 256; int col = (int)(t % N); int b = (int)(t / N);
    out[i] = (c < 128) ? keyT[((size_t)b * N + col) * 128 + c]
                       : qT[((size_t)b * N + col) * 128 + (c - 128)];
}

__global__ void posrel_kernel(const float* __restrict__ pos, const int* __restrict__ idx,
                              unsigned short* __restrict__ out, int N, int NKC)
{
    size_t i = (size_t)blockIdx.x * 256 + threadIdx.x;     // out (B,NKC,32)
    size_t total = (size_t)2 * NKC * 32;
    if (i >= total) return;
    int c = (int)(i % 32); size_t t = i / 32; int nk = (int)(t % NKC); int b = (int)(t / NKC);
    unsigned short v = 0;
    if (c < 3) {
        int n = nk >> 4, k = nk & 15;
        int j = idx[((size_t)b * N + n) * 16 + k];
        v = (unsigned short)f2bf(pos[((size_t)b * 3 + c) * N + n] - pos[((size_t)b * 3 + c) * N + j]);
    }
    out[i] = v;
}

__global__ void attnin_kernel(const unsigned short* __restrict__ qf, const unsigned short* __restrict__ kf,
                              const unsigned short* __restrict__ pe, const int* __restrict__ idx,
                              unsigned short* __restrict__ out, int N, int NKC)
{
    size_t i = (size_t)blockIdx.x * 256 + threadIdx.x;     // out (B,NKC,64)
    size_t total = (size_t)2 * NKC * 64;
    if (i >= total) return;
    int c = (int)(i % 64); size_t t = i / 64; int nk = (int)(t % NKC); int b = (int)(t / NKC);
    int n = nk >> 4, k = nk & 15;
    int j = idx[((size_t)b * N + n) * 16 + k];
    float v = bf2f(qf[((size_t)b * N + n) * 64 + c]) - bf2f(kf[((size_t)b * N + j) * 64 + c]) + bf2f(pe[i]);
    out[i] = (unsigned short)f2bf(v);
}

__global__ void softagg_kernel(const unsigned short* __restrict__ logits, const unsigned short* __restrict__ pe,
                               const unsigned short* __restrict__ vf, unsigned short* __restrict__ agg,
                               int N, int NKC)
{
    size_t i = (size_t)blockIdx.x * 256 + threadIdx.x;     // thread per (b,n,c), c<64
    size_t total = (size_t)2 * N * 64;
    if (i >= total) return;
    int c = (int)(i % 64); size_t t = i / 64; int n = (int)(t % N); int b = (int)(t / N);
    size_t base = ((size_t)b * NKC + (size_t)n * 16) * 64 + c;
    float L[16];
#pragma unroll
    for (int k = 0; k < 16; ++k) L[k] = bf2f(logits[base + (size_t)k * 64]);
    float mx = L[0];
#pragma unroll
    for (int k = 1; k < 16; ++k) mx = fmaxf(mx, L[k]);
    float v0 = bf2f(vf[((size_t)b * N + n) * 64 + c]);
    float s = 0.f, acc = 0.f;
#pragma unroll
    for (int k = 0; k < 16; ++k) {
        float e = __expf(L[k] - mx);
        s += e;
        acc += e * (v0 + bf2f(pe[base + (size_t)k * 64]));
    }
    agg[((size_t)b * N + n) * 64 + c] = (unsigned short)f2bf(acc / s);
}

// ConvTranspose1d(32,128,k=2,stride=2): out[b,2n+k][o] = sum_i fc[b,n][i]*w[i,o,k]
__global__ void psup_kernel(const unsigned short* __restrict__ fc, const float* __restrict__ w,
                            unsigned short* __restrict__ out, int N)
{
    size_t i = (size_t)blockIdx.x * 256 + threadIdx.x;     // thread per (b,n,o)
    size_t total = (size_t)2 * N * 128;
    if (i >= total) return;
    int o = (int)(i % 128); size_t t = i / 128; int n = (int)(t % N); int b = (int)(t / N);
    float s0 = 0.f, s1 = 0.f;
#pragma unroll 4
    for (int ci = 0; ci < 32; ++ci) {
        float f = bf2f(fc[((size_t)b * N + n) * 32 + ci]);
        s0 += f * w[ci * 256 + o * 2 + 0];
        s1 += f * w[ci * 256 + o * 2 + 1];
    }
    size_t N2 = (size_t)2 * N;
    out[((size_t)b * N2 + 2 * n)     * 128 + o] = (unsigned short)f2bf(s0);
    out[((size_t)b * N2 + 2 * n + 1) * 128 + o] = (unsigned short)f2bf(s1);
}

__global__ void dfcat_kernel(const unsigned short* __restrict__ ps, const unsigned short* __restrict__ H,
                             unsigned short* __restrict__ out, int N2)
{
    size_t i = (size_t)blockIdx.x * 256 + threadIdx.x;     // out (B,N2,256)
    size_t total = (size_t)2 * N2 * 256;
    if (i >= total) return;
    int c = (int)(i % 256); size_t t = i / 256; int m = (int)(t % N2); int b = (int)(t / N2);
    out[i] = (c < 128) ? ps[((size_t)b * N2 + m) * 128 + c]
                       : H[((size_t)b * (N2 / 2) + (m >> 1)) * 128 + (c - 128)];
}

__global__ void child_kernel(const float* __restrict__ pcd, const unsigned short* __restrict__ delta,
                             float* __restrict__ out, int N)
{
    size_t i = (size_t)blockIdx.x * 256 + threadIdx.x;     // out (B,3,2N) f32
    size_t total = (size_t)2 * 3 * (2 * N);
    if (i >= total) return;
    int N2 = 2 * N;
    int m = (int)(i % N2); size_t t = i / N2; int c = (int)(t % 3); int b = (int)(t / 3);
    out[i] = pcd[((size_t)b * 3 + c) * N + (m >> 1)] + bf2f(delta[((size_t)b * N2 + m) * 16 + c]);
}

__global__ void bnprep_kernel(const float* g, const float* be, const float* mean,
                              const float* var, float* sc, float* sh, int C)
{
    int i = blockIdx.x * 256 + threadIdx.x;
    if (i >= C) return;
    float s = g[i] * rsqrtf(var[i] + 1e-5f);
    sc[i] = s;
    sh[i] = be[i] - mean[i] * s;
}

// ---------------------------------------------------------------------------
// Host-side orchestration
// ---------------------------------------------------------------------------
static inline void launch_gemm(hipStream_t s,
    const unsigned short* Xt, const unsigned int* WP,
    const float* bias, const float* bnsc, const float* bnsh,
    const unsigned short* resT, unsigned short* Yt, float* Yf,
    int Cp, int O, int Op, int colsIn, int colInBase,
    int colsOut, int colOutBase, int colCount, int NfOut, int actOut, int reluT)
{
    int nSR = (O + 15) / 16;
    dim3 grid((colCount + 127) / 128, (O + 63) / 64, 2);
    gemm_wmma2<<<grid, 256, 0, s>>>(Xt, WP, bias, bnsc, bnsh, resT, Yt, Yf,
                                    Cp, O, Op, nSR, colsIn, colInBase,
                                    colsOut, colOutBase, colCount, NfOut, actOut, reluT);
}
static inline dim3 g1(size_t total) { return dim3((unsigned)((total + 255) / 256)); }

extern "C" void kernel_launch(void* const* d_in, const int* in_sizes, int n_in,
                              void* d_out, int out_size, void* d_ws, size_t ws_size,
                              hipStream_t stream)
{
    (void)in_sizes; (void)n_in; (void)out_size; (void)ws_size;
    const int N = 8192, N2 = 16384, NKC = 131072;

    // ---- inputs (pytree leaf order: tensors, then params with sorted dict keys) ----
    const float* pcd   = (const float*)d_in[0];
    const float* fg    = (const float*)d_in[1];
    const float* Kprev = (const float*)d_in[2];
    const float* m1w1 = (const float*)d_in[3];   const float* m1b1 = (const float*)d_in[4];
    const float* m1w2 = (const float*)d_in[5];   const float* m1b2 = (const float*)d_in[6];
    const float* m2w1 = (const float*)d_in[7];   const float* m2b1 = (const float*)d_in[8];
    const float* m2w2 = (const float*)d_in[9];   const float* m2b2 = (const float*)d_in[10];
    const float* dlw1 = (const float*)d_in[11];  const float* dlb1 = (const float*)d_in[12];
    const float* dlw2 = (const float*)d_in[13];  const float* dlb2 = (const float*)d_in[14];
    const float* dfb1 = (const float*)d_in[15];  const float* dfb2 = (const float*)d_in[16];
    const float* dfbs = (const float*)d_in[17];  const float* dfw1 = (const float*)d_in[18];
    const float* dfw2 = (const float*)d_in[19];  const float* dfws = (const float*)d_in[20];
    const float* psw1 = (const float*)d_in[21];  const float* psb1 = (const float*)d_in[22];
    const float* psw2 = (const float*)d_in[23];  const float* psb2 = (const float*)d_in[24];
    const float* ps_w = (const float*)d_in[25];
    const float* ab1  = (const float*)d_in[26];  const float* ab2  = (const float*)d_in[27];
    const float* abe  = (const float*)d_in[28];  const float* ag   = (const float*)d_in[29];
    const float* am   = (const float*)d_in[30];  const float* av   = (const float*)d_in[31];
    const float* aw1  = (const float*)d_in[32];  const float* aw2  = (const float*)d_in[33];
    const float* be_  = (const float*)d_in[34];  const float* bk   = (const float*)d_in[35];
    const float* bq   = (const float*)d_in[36];  const float* bv   = (const float*)d_in[37];
    const float* vb1  = (const float*)d_in[38];  const float* vb2  = (const float*)d_in[39];
    const float* vbs  = (const float*)d_in[40];  const float* vw1  = (const float*)d_in[41];
    const float* vw2  = (const float*)d_in[42];  const float* vws  = (const float*)d_in[43];
    const float* pb1  = (const float*)d_in[44];  const float* pb2  = (const float*)d_in[45];
    const float* pbe  = (const float*)d_in[46];  const float* pg   = (const float*)d_in[47];
    const float* pm   = (const float*)d_in[48];  const float* pv   = (const float*)d_in[49];
    const float* pw1  = (const float*)d_in[50];  const float* pw2  = (const float*)d_in[51];
    const float* we   = (const float*)d_in[52];  const float* wk   = (const float*)d_in[53];
    const float* wq   = (const float*)d_in[54];  const float* wv   = (const float*)d_in[55];

    // ---- deterministic bump allocator over workspace ----
    char* p = (char*)d_ws;
    auto A = [&](size_t bytes) { char* r = p; p += (bytes + 255) & ~(size_t)255; return r; };
    auto wsz = [](int C, int O) { return (size_t)((C + 31) / 32) * ((O + 15) / 16) * 256 * 4; };

    unsigned int* wpm1w1 = (unsigned int*)A(wsz(3, 64));
    unsigned int* wpm1w2 = (unsigned int*)A(wsz(64, 128));
    unsigned int* wpm2w1 = (unsigned int*)A(wsz(768, 256));
    unsigned int* wpm2w2 = (unsigned int*)A(wsz(256, 128));
    unsigned int* wpvw1  = (unsigned int*)A(wsz(256, 128));
    unsigned int* wpvw2  = (unsigned int*)A(wsz(128, 128));
    unsigned int* wpvws  = (unsigned int*)A(wsz(256, 128));
    unsigned int* wpwk   = (unsigned int*)A(wsz(128, 64));
    unsigned int* wpwq   = (unsigned int*)A(wsz(128, 64));
    unsigned int* wpwv   = (unsigned int*)A(wsz(128, 64));
    unsigned int* wppw1  = (unsigned int*)A(wsz(3, 64));
    unsigned int* wppw2  = (unsigned int*)A(wsz(64, 64));
    unsigned int* wpaw1  = (unsigned int*)A(wsz(64, 256));
    unsigned int* wpaw2  = (unsigned int*)A(wsz(256, 64));
    unsigned int* wpwe   = (unsigned int*)A(wsz(64, 128));
    unsigned int* wppsw1 = (unsigned int*)A(wsz(128, 64));
    unsigned int* wppsw2 = (unsigned int*)A(wsz(64, 32));
    unsigned int* wpdfw1 = (unsigned int*)A(wsz(256, 128));
    unsigned int* wpdfw2 = (unsigned int*)A(wsz(128, 128));
    unsigned int* wpdfws = (unsigned int*)A(wsz(256, 128));
    unsigned int* wpdlw1 = (unsigned int*)A(wsz(128, 64));
    unsigned int* wpdlw2 = (unsigned int*)A(wsz(64, 3));

    typedef unsigned short us;
    us* pcdT = (us*)A((size_t)2 * N * 32 * 2);
    us* KprT = (us*)A((size_t)2 * N * 128 * 2);
    us* t64b = (us*)A((size_t)2 * N * 64 * 2);
    us* f128 = (us*)A((size_t)2 * N * 128 * 2);
    us* f1c  = (us*)A((size_t)2 * N * 768 * 2);
    us* q256 = (us*)A((size_t)2 * N * 256 * 2);
    us* Qb   = (us*)A((size_t)2 * N * 128 * 2);
    us* kqb  = (us*)A((size_t)2 * N * 256 * 2);
    us* vtmp = (us*)A((size_t)2 * N * 128 * 2);
    us* valb = (us*)A((size_t)2 * N * 128 * 2);
    us* kfb  = (us*)A((size_t)2 * N * 64 * 2);
    us* qfb  = (us*)A((size_t)2 * N * 64 * 2);
    us* vfb  = (us*)A((size_t)2 * N * 64 * 2);
    int* idxb = (int*)A((size_t)2 * N * 16 * 4);
    us* prT  = (us*)A((size_t)2 * NKC * 32 * 2);
    us* peb  = (us*)A((size_t)2 * NKC * 64 * 2);
    us* atb  = (us*)A((size_t)2 * NKC * 64 * 2);
    us* amid = (us*)A((size_t)2 * 8192 * 256 * 2);
    us* aggb = (us*)A((size_t)2 * N * 64 * 2);
    us* Hbuf = (us*)A((size_t)2 * N * 128 * 2);
    us* fc64 = (us*)A((size_t)2 * N * 64 * 2);
    us* fc32 = (us*)A((size_t)2 * N * 32 * 2);
    us* pso  = (us*)A((size_t)2 * N2 * 128 * 2);
    us* dfc  = (us*)A((size_t)2 * N2 * 256 * 2);
    us* dft  = (us*)A((size_t)2 * N2 * 128 * 2);
    us* Kpart= (us*)A((size_t)2 * N2 * 128 * 2);
    us* KcT  = (us*)A((size_t)2 * N2 * 128 * 2);
    us* d64  = (us*)A((size_t)2 * N2 * 64 * 2);
    us* delt = (us*)A((size_t)2 * N2 * 16 * 2);
    float* gmaxb = (float*)A(2 * 128 * 4);
    float* pesc = (float*)A(64 * 4);  float* pesh = (float*)A(64 * 4);
    float* asc  = (float*)A(256 * 4); float* ash  = (float*)A(256 * 4);

    float* KcF   = (float*)d_out + 98304;   // K_curr (B,128,N2) f32, reference layout
    float* child = (float*)d_out;           // pcd_child (B,3,N2) f32

    // ---- one-time weight packs + input transposes + BN prep ----
    auto wpack = [&](const float* w, unsigned int* o, int C, int O) {
        int nks = (C + 31) / 32, nsr = (O + 15) / 16, total = nks * nsr * 256;
        wpack_kernel<<<dim3((total + 255) / 256), 256, 0, stream>>>(w, o, C, O, nks, nsr);
    };
    wpack(m1w1, wpm1w1, 3, 64);     wpack(m1w2, wpm1w2, 64, 128);
    wpack(m2w1, wpm2w1, 768, 256);  wpack(m2w2, wpm2w2, 256, 128);
    wpack(vw1, wpvw1, 256, 128);    wpack(vw2, wpvw2, 128, 128);   wpack(vws, wpvws, 256, 128);
    wpack(wk, wpwk, 128, 64);       wpack(wq, wpwq, 128, 64);      wpack(wv, wpwv, 128, 64);
    wpack(pw1, wppw1, 3, 64);       wpack(pw2, wppw2, 64, 64);
    wpack(aw1, wpaw1, 64, 256);     wpack(aw2, wpaw2, 256, 64);
    wpack(we, wpwe, 64, 128);
    wpack(psw1, wppsw1, 128, 64);   wpack(psw2, wppsw2, 64, 32);
    wpack(dfw1, wpdfw1, 256, 128);  wpack(dfw2, wpdfw2, 128, 128); wpack(dfws, wpdfws, 256, 128);
    wpack(dlw1, wpdlw1, 128, 64);   wpack(dlw2, wpdlw2, 64, 3);
    trpack_kernel<<<g1((size_t)2 * N * 32), 256, 0, stream>>>(pcd, pcdT, 3, 32, N);
    trpack_kernel<<<g1((size_t)2 * N * 128), 256, 0, stream>>>(Kprev, KprT, 128, 128, N);
    bnprep_kernel<<<1, 256, 0, stream>>>(pg, pbe, pm, pv, pesc, pesh, 64);
    bnprep_kernel<<<1, 256, 0, stream>>>(ag, abe, am, av, asc, ash, 256);

    // ================= mlp1 =================
    launch_gemm(stream, pcdT, wpm1w1, m1b1, 0, 0, 0, t64b, 0, 32, 64, 64,  N, 0, N, 0, N, 0, 1, 0);
    launch_gemm(stream, t64b, wpm1w2, m1b2, 0, 0, 0, f128, 0, 64, 128, 128, N, 0, N, 0, N, 0, 0, 0);
    colmax_kernel<<<dim3(2), 128, 0, stream>>>(f128, gmaxb, N, 128);
    f1cat_kernel<<<g1((size_t)2 * N * 768), 256, 0, stream>>>(f128, gmaxb, fg, f1c, N);

    // ================= mlp2 -> Q =================
    launch_gemm(stream, f1c,  wpm2w1, m2b1, 0, 0, 0, q256, 0, 768, 256, 256, N, 0, N, 0, N, 0, 1, 0);
    launch_gemm(stream, q256, wpm2w2, m2b2, 0, 0, 0, Qb,   0, 256, 128, 128, N, 0, N, 0, N, 0, 0, 0);

    // ================= value = mlp_res(concat(key,Q)) =================
    kqcat_kernel<<<g1((size_t)2 * N * 256), 256, 0, stream>>>(KprT, Qb, kqb, N);
    launch_gemm(stream, kqb,  wpvw1, vb1, 0, 0, 0,    vtmp, 0, 256, 128, 128, N, 0, N, 0, N, 0, 1, 0);
    launch_gemm(stream, vtmp, wpvw2, vb2, 0, 0, 0,    valb, 0, 128, 128, 128, N, 0, N, 0, N, 0, 0, 0);
    launch_gemm(stream, kqb,  wpvws, vbs, 0, 0, valb, valb, 0, 256, 128, 128, N, 0, N, 0, N, 0, 0, 0);

    // ================= k_f / q_f / v_f =================
    launch_gemm(stream, KprT, wpwk, bk, 0, 0, 0, kfb, 0, 128, 64, 64, N, 0, N, 0, N, 0, 0, 0);
    launch_gemm(stream, Qb,   wpwq, bq, 0, 0, 0, qfb, 0, 128, 64, 64, N, 0, N, 0, N, 0, 0, 0);
    launch_gemm(stream, valb, wpwv, bv, 0, 0, 0, vfb, 0, 128, 64, 64, N, 0, N, 0, N, 0, 0, 0);

    // ================= KNN + gathers =================
    knn_kernel<<<dim3(N / 256, 2), 256, 0, stream>>>(pcd, idxb, N);
    posrel_kernel<<<g1((size_t)2 * NKC * 32), 256, 0, stream>>>(pcd, idxb, prT, N, NKC);

    // ================= positional encoding (3->64 bn relu ->64, in place) ======
    launch_gemm(stream, prT, wppw1, pb1, pesc, pesh, 0, peb, 0, 32, 64, 64, NKC, 0, NKC, 0, NKC, 0, 1, 0);
    launch_gemm(stream, peb, wppw2, pb2, 0, 0, 0,       peb, 0, 64, 64, 64, NKC, 0, NKC, 0, NKC, 0, 0, 0);

    // ================= attention MLP 64->256(bn relu)->64, chunked in-place ====
    attnin_kernel<<<g1((size_t)2 * NKC * 64), 256, 0, stream>>>(qfb, kfb, peb, idxb, atb, N, NKC);
    const int CH = 8192;
    for (int cs = 0; cs < NKC; cs += CH) {
        launch_gemm(stream, atb,  wpaw1, ab1, asc, ash, 0, amid, 0, 64, 256, 256, NKC, cs, CH, 0, CH, 0, 1, 0);
        launch_gemm(stream, amid, wpaw2, ab2, 0, 0, 0,     atb,  0, 256, 64, 64, CH, 0, NKC, cs, CH, 0, 0, 0);
    }

    // ================= softmax over neighbors + aggregate, then H ==============
    softagg_kernel<<<g1((size_t)2 * N * 64), 256, 0, stream>>>(atb, peb, vfb, aggb, N, NKC);
    launch_gemm(stream, aggb, wpwe, be_, 0, 0, valb, Hbuf, 0, 64, 128, 128, N, 0, N, 0, N, 0, 0, 0);

    // ================= mlp_ps + transposed conv upsample =======================
    launch_gemm(stream, Hbuf, wppsw1, psb1, 0, 0, 0, fc64, 0, 128, 64, 64, N, 0, N, 0, N, 0, 1, 0);
    launch_gemm(stream, fc64, wppsw2, psb2, 0, 0, 0, fc32, 0, 64, 32, 32, N, 0, N, 0, N, 0, 0, 0);
    psup_kernel<<<g1((size_t)2 * N * 128), 256, 0, stream>>>(fc32, ps_w, pso, N);

    // ================= K_curr = mlp_res(concat(fc_up, H_up)), dual write =======
    dfcat_kernel<<<g1((size_t)2 * N2 * 256), 256, 0, stream>>>(pso, Hbuf, dfc, N2);
    launch_gemm(stream, dfc, wpdfw1, dfb1, 0, 0, 0,     dft,   0,  256, 128, 128, N2, 0, N2, 0, N2, 0, 1, 0);
    launch_gemm(stream, dft, wpdfw2, dfb2, 0, 0, 0,     Kpart, 0,  128, 128, 128, N2, 0, N2, 0, N2, 0, 0, 0);
    launch_gemm(stream, dfc, wpdfws, dfbs, 0, 0, Kpart, KcT,   KcF, 256, 128, 128, N2, 0, N2, 0, N2, N2, 0, 1);

    // ================= delta = tanh(mlp(relu(K_curr))); pcd_child ==============
    launch_gemm(stream, KcT, wpdlw1, dlb1, 0, 0, 0, d64,  0, 128, 64, 64, N2, 0, N2, 0, N2, 0, 1, 0);
    launch_gemm(stream, d64, wpdlw2, dlb2, 0, 0, 0, delt, 0, 64,  3,  16, N2, 0, N2, 0, N2, 0, 2, 0);
    child_kernel<<<g1((size_t)2 * 3 * N2), 256, 0, stream>>>(pcd, delt, child, N);
}
